// AdaINResBlock1_78932908966601
// MI455X (gfx1250) — matrix operand order repaired and checked by reading, not compiled, not run on hardware
//
#include <hip/hip_runtime.h>
#include <hip/hip_bf16.h>
#include <stdint.h>

#define BB 8
#define TT 4096
#define CC 512
#define SS 64
#define KK 3
#define EPSV 1e-5f
#define TCHUNK 8
#define TSTEP (TT / TCHUNK)   // 512
#define TM 128
#define TN 128
#define TKS 32
#define NKIT (KK * CC / TKS)  // 48

typedef __bf16 bf16_t;
typedef bf16_t bf16x16 __attribute__((ext_vector_type(16)));
typedef float v8f __attribute__((ext_vector_type(8)));

__device__ __forceinline__ unsigned short f2bf(float f) {
  uint32_t u = __float_as_uint(f);
  uint32_t r = ((u >> 16) & 1u) + 0x7FFFu;   // round-to-nearest-even
  return (unsigned short)((u + r) >> 16);
}

// ---------------------------------------------------------------- style FC
// styles[(i*2+f)][b][0..1023] = s[b] @ fc_w[i] + fc_b[i]
__global__ void style_kernel(const float* __restrict__ s,
                             const float* __restrict__ fc1_w, const float* __restrict__ fc1_b,
                             const float* __restrict__ fc2_w, const float* __restrict__ fc2_b,
                             float* __restrict__ styles) {
  int q = blockIdx.x;            // 0..47
  int b = q % BB;
  int fi = q / BB;               // 0..5 == i*2+f
  int i = fi >> 1, f = fi & 1;
  const float* W    = (f == 0 ? fc1_w : fc2_w) + (size_t)i * SS * 2 * CC;
  const float* bias = (f == 0 ? fc1_b : fc2_b) + (size_t)i * 2 * CC;
  __shared__ float sv[SS];
  if (threadIdx.x < SS) sv[threadIdx.x] = s[b * SS + threadIdx.x];
  __syncthreads();
  for (int j = threadIdx.x; j < 2 * CC; j += blockDim.x) {
    float acc = bias[j];
#pragma unroll 8
    for (int k = 0; k < SS; ++k) acc += sv[k] * W[(size_t)k * 2 * CC + j];
    styles[((size_t)fi * BB + b) * (2 * CC) + j] = acc;
  }
}

// ---------------------------------------------------------------- instance-norm stats
// grid (TCHUNK, BB), 128 threads; each thread reduces 4 channels over TSTEP steps
__global__ void stats_partial_kernel(const float* __restrict__ X,
                                     float* __restrict__ psum, float* __restrict__ psumsq) {
  int c4 = threadIdx.x;          // float4 channel index: covers all 512 channels
  int tc = blockIdx.x;
  int b  = blockIdx.y;
  const float4* p = (const float4*)(X + ((size_t)b * TT + (size_t)tc * TSTEP) * CC) + c4;
  float4 s  = {0.f, 0.f, 0.f, 0.f};
  float4 s2 = {0.f, 0.f, 0.f, 0.f};
  for (int t = 0; t < TSTEP; ++t) {
    float4 v = p[(size_t)t * (CC / 4)];
    s.x += v.x; s.y += v.y; s.z += v.z; s.w += v.w;
    s2.x += v.x * v.x; s2.y += v.y * v.y; s2.z += v.z * v.z; s2.w += v.w * v.w;
  }
  size_t o = ((size_t)tc * BB + b) * (CC / 4) + c4;
  ((float4*)psum)[o] = s;
  ((float4*)psumsq)[o] = s2;
}

__global__ void stats_final_kernel(const float* __restrict__ psum, const float* __restrict__ psumsq,
                                   float* __restrict__ mean, float* __restrict__ rstd) {
  int idx = blockIdx.x * 256 + threadIdx.x;   // b*CC + c
  int b = idx / CC, c = idx % CC;
  float s = 0.f, s2 = 0.f;
#pragma unroll
  for (int tc = 0; tc < TCHUNK; ++tc) {
    size_t o = ((size_t)tc * BB + b) * CC + c;
    s += psum[o]; s2 += psumsq[o];
  }
  float m   = s * (1.0f / TT);
  float var = fmaxf(s2 * (1.0f / TT) - m * m, 0.f);
  mean[idx] = m;
  rstd[idx] = rsqrtf(var + EPSV);
}

// ---------------------------------------------------------------- fused adain + snake + bf16 cast
__global__ void adain_snake_kernel(const float* __restrict__ X,
                                   const float* __restrict__ mean, const float* __restrict__ rstd,
                                   const float* __restrict__ style,   // [B][2C]
                                   const float* __restrict__ alpha,   // [C]
                                   unsigned short* __restrict__ act) {
  size_t f4   = (size_t)blockIdx.x * blockDim.x + threadIdx.x;  // float4 index
  size_t flat = f4 * 4;
  int b = (int)(flat / ((size_t)TT * CC));
  int c = (int)(flat % CC);
  const float4 xv = ((const float4*)X)[f4];
  float in[4] = {xv.x, xv.y, xv.z, xv.w};
  unsigned short h[4];
#pragma unroll
  for (int j = 0; j < 4; ++j) {
    int cj = c + j;
    int bc = b * CC + cj;
    float g  = style[(size_t)b * 2 * CC + cj];
    float be = style[(size_t)b * 2 * CC + CC + cj];
    float a  = alpha[cj];
    float xn = (in[j] - mean[bc]) * rstd[bc];
    float y  = (1.0f + g) * xn + be;
    float sn = __sinf(a * y);
    h[j] = f2bf(y + sn * sn / a);
  }
  uint2 o;
  o.x = (uint32_t)h[0] | ((uint32_t)h[1] << 16);
  o.y = (uint32_t)h[2] | ((uint32_t)h[3] << 16);
  ((uint2*)act)[f4] = o;
}

// ---------------------------------------------------------------- weight-norm prep
__global__ void wnorm_kernel(const float* __restrict__ v, const float* __restrict__ g,
                             float* __restrict__ wscale) {
  int co = blockIdx.x * blockDim.x + threadIdx.x;   // 0..511
  float s = 0.f;
  for (int e = 0; e < KK * CC; ++e) {
    float w = v[(size_t)e * CC + co];
    s += w * w;
  }
  wscale[co] = g[co] * rsqrtf(s);
}

// wqT[co][e] = bf16( v[e][co] * wscale[co] ),  e = k*CC+ci  (transposed, WMMA-B friendly)
__global__ void wconvert_kernel(const float* __restrict__ v, const float* __restrict__ wscale,
                                unsigned short* __restrict__ wqT) {
  size_t flat = (size_t)blockIdx.x * blockDim.x + threadIdx.x;   // co*(K*C) + e
  int co = (int)(flat / (KK * CC));
  int e  = (int)(flat % (KK * CC));
  wqT[flat] = f2bf(v[(size_t)e * CC + co] * wscale[co]);
}

// ---------------------------------------------------------------- conv as implicit GEMM via WMMA
// y[b,t,co] = bias[co] + sum_{k,ci} act_bf16[b, t+(k-1)*dil, ci] * wqT_bf16[co, k*C+ci] (+res)
// Tiles staged with async global->LDS DMA (ASYNCcnt), double-buffered.
__global__ __launch_bounds__(256)
void conv_wmma_kernel(const unsigned short* __restrict__ act,  // bf16 [B][T][C]
                      const unsigned short* __restrict__ wqT,  // bf16 [C][3C]
                      const float* __restrict__ bias,          // [C]
                      const float* res,                        // [B][T][C] or null
                      float* out,                              // [B][T][C]
                      const unsigned short* __restrict__ zeroPad,  // >=64B of zeros
                      int dil) {
  __shared__ unsigned short As[2][TM * TKS];   // [buf][t_local][k]  2 x 8 KB
  __shared__ unsigned short Bs[2][TN * TKS];   // [buf][n_local][k]  2 x 8 KB

  const int tid  = threadIdx.x;
  const int lane = tid & 31;
  const int wave = tid >> 5;
  const int half = lane >> 4;    // lane group 0..1
  const int l16  = lane & 15;
  const int wm   = wave & 3;     // 4 M-subtiles of 32
  const int wn   = wave >> 2;    // 2 N-subtiles of 64
  const int t0   = blockIdx.x * TM;
  const int n0   = blockIdx.y * TN;
  const int b    = blockIdx.z;

  const int r  = tid >> 1;   // 0..127: tile row (A: time, B: cout)
  const int hh = tid & 1;    // half-row (16 bf16 = 32 bytes each)

  v8f acc[2][4];
  {
    v8f zero = {0.f, 0.f, 0.f, 0.f, 0.f, 0.f, 0.f, 0.f};
#pragma unroll
    for (int mi = 0; mi < 2; ++mi)
#pragma unroll
      for (int ni = 0; ni < 4; ++ni) acc[mi][ni] = zero;
  }

  const unsigned short* bRow = wqT + (size_t)(n0 + r) * (KK * CC) + hh * 16;

  // LDS destination byte offsets (workgroup-relative, low 32 bits of generic ptr)
  uint32_t ldsA[2], ldsB[2];
#pragma unroll
  for (int u = 0; u < 2; ++u) {
    ldsA[u] = (uint32_t)(uintptr_t)&As[u][r * TKS + hh * 16];
    ldsB[u] = (uint32_t)(uintptr_t)&Bs[u][r * TKS + hh * 16];
  }

  // ---- async tile stager: global -> LDS, 4 x b128 per thread, EXEC stays all-ones
  auto issueTiles = [&](int kt, int buf) {
    const int k0  = kt * TKS;
    const int tap = k0 / CC;
    const int ci0 = k0 % CC;
    int t_in = t0 + r + (tap - 1) * dil;
    const unsigned short* srcA =
        ((unsigned)t_in < (unsigned)TT)
            ? act + ((size_t)b * TT + t_in) * CC + ci0 + hh * 16
            : zeroPad;                                   // branchless zero padding
    uint64_t ga = (uint64_t)(uintptr_t)srcA;
    uint64_t gb = (uint64_t)(uintptr_t)(bRow + k0);
    uint32_t la = ldsA[buf];
    uint32_t lb = ldsB[buf];
    asm volatile("global_load_async_to_lds_b128 %0, %1, off"
                 :: "v"(la), "v"(ga) : "memory");
    asm volatile("global_load_async_to_lds_b128 %0, %1, off offset:16"
                 :: "v"(la), "v"(ga) : "memory");
    asm volatile("global_load_async_to_lds_b128 %0, %1, off"
                 :: "v"(lb), "v"(gb) : "memory");
    asm volatile("global_load_async_to_lds_b128 %0, %1, off offset:16"
                 :: "v"(lb), "v"(gb) : "memory");
  };

  issueTiles(0, 0);

  union Frag { bf16x16 v; uint4 q[2]; };

  for (int kt = 0; kt < NKIT; ++kt) {
    // my async writes into buf (kt&1) complete, then all waves rendezvous
    asm volatile("s_wait_asynccnt 0x0" ::: "memory");
    __syncthreads();

    const bool more = (kt + 1) < NKIT;
    if (more) {
      issueTiles(kt + 1, (kt + 1) & 1);   // DMA next tile while we compute
      if (kt + 2 < NKIT) {                // warm GL2 two tiles ahead
        const int k2  = (kt + 2) * TKS;
        const int tap = k2 / CC;
        int t_in = t0 + r + (tap - 1) * dil;
        if ((unsigned)t_in < (unsigned)TT)
          __builtin_prefetch(act + ((size_t)b * TT + t_in) * CC + (k2 % CC), 0, 0);
      }
    }

    // ---- fragments from LDS (documented 16-bit A 16x32 / B 32x16 lane layouts)
    const int buf = kt & 1;
    Frag a[2], bf[4];
#pragma unroll
    for (int mi = 0; mi < 2; ++mi) {
      int m = wm * 32 + mi * 16 + l16;
      const uint4* p = (const uint4*)(&As[buf][m * TKS]);
      a[mi].q[0] = p[half];          // K = 8*half .. +8
      a[mi].q[1] = p[2 + half];      // K = 16+8*half .. +8
    }
#pragma unroll
    for (int ni = 0; ni < 4; ++ni) {
      int n = wn * 64 + ni * 16 + l16;
      const uint4* p = (const uint4*)(&Bs[buf][n * TKS]);
      bf[ni].q[0] = p[half];         // K = 8*half .. +8
      bf[ni].q[1] = p[2 + half];     // K = 16+8*half .. +8
    }
#pragma unroll
    for (int mi = 0; mi < 2; ++mi)
#pragma unroll
      for (int ni = 0; ni < 4; ++ni)
        acc[mi][ni] = __builtin_amdgcn_wmma_f32_16x16x32_bf16(
            false, a[mi].v, false, bf[ni].v, (short)0, acc[mi][ni], false, false);
  }

  // ---- epilogue: bias (+ residual) and store
#pragma unroll
  for (int mi = 0; mi < 2; ++mi) {
#pragma unroll
    for (int ni = 0; ni < 4; ++ni) {
      int cn = n0 + wn * 64 + ni * 16 + l16;
      float bv = bias[cn];
#pragma unroll
      for (int v = 0; v < 8; ++v) {
        int tt = t0 + wm * 32 + mi * 16 + v + 8 * half;   // C/D: M = vgpr + 8*half
        size_t o = ((size_t)b * TT + tt) * CC + cn;
        float val = acc[mi][ni][v] + bv;
        if (res) val += res[o];
        out[o] = val;
      }
    }
  }
}

// ---------------------------------------------------------------- host
extern "C" void kernel_launch(void* const* d_in, const int* in_sizes, int n_in,
                              void* d_out, int out_size, void* d_ws, size_t ws_size,
                              hipStream_t stream) {
  const float* x       = (const float*)d_in[0];
  const float* s       = (const float*)d_in[1];
  const float* fc1_w   = (const float*)d_in[2];
  const float* fc1_b   = (const float*)d_in[3];
  const float* alpha1  = (const float*)d_in[4];
  const float* conv1_v = (const float*)d_in[5];
  const float* conv1_g = (const float*)d_in[6];
  const float* conv1_b = (const float*)d_in[7];
  const float* fc2_w   = (const float*)d_in[8];
  const float* fc2_b   = (const float*)d_in[9];
  const float* alpha2  = (const float*)d_in[10];
  const float* conv2_v = (const float*)d_in[11];
  const float* conv2_g = (const float*)d_in[12];
  const float* conv2_b = (const float*)d_in[13];
  float* X = (float*)d_out;            // running residual state

  char* w = (char*)d_ws;
  size_t off = 0;
  auto carve = [&](size_t bytes) -> char* {
    char* p = w + off;
    off += (bytes + 255) & ~(size_t)255;
    return p;
  };
  unsigned short* actB   = (unsigned short*)carve((size_t)BB * TT * CC * 2);  // 32 MB
  float*          Y      = (float*)carve((size_t)BB * TT * CC * 4);           // 64 MB
  unsigned short* wqT    = (unsigned short*)carve((size_t)KK * CC * CC * 2);  // 1.5 MB
  float*          wscale = (float*)carve((size_t)CC * 4);
  float*          styles = (float*)carve((size_t)6 * BB * 2 * CC * 4);
  float*          psum   = (float*)carve((size_t)TCHUNK * BB * CC * 4);
  float*          psumsq = (float*)carve((size_t)TCHUNK * BB * CC * 4);
  float*          meanB  = (float*)carve((size_t)BB * CC * 4);
  float*          rstdB  = (float*)carve((size_t)BB * CC * 4);
  unsigned short* zeroP  = (unsigned short*)carve(256);                       // zero pad source

  hipMemsetAsync(zeroP, 0, 256, stream);
  hipMemcpyAsync(X, x, (size_t)BB * TT * CC * 4, hipMemcpyDeviceToDevice, stream);

  style_kernel<<<6 * BB, 256, 0, stream>>>(s, fc1_w, fc1_b, fc2_w, fc2_b, styles);

  const int dils[3] = {1, 3, 5};
  const dim3 statsGrid(TCHUNK, BB);
  const dim3 convGrid(TT / TM, CC / TN, BB);
  const int elemBlocks = (int)((size_t)BB * TT * CC / 4 / 256);

  for (int i = 0; i < 3; ++i) {
    const float* v1 = conv1_v + (size_t)i * KK * CC * CC;
    const float* v2 = conv2_v + (size_t)i * KK * CC * CC;

    // --- adain(X) -> snake -> conv1 (dilated) -> Y
    wnorm_kernel<<<CC / 256, 256, 0, stream>>>(v1, conv1_g + i * CC, wscale);
    wconvert_kernel<<<KK * CC * CC / 256, 256, 0, stream>>>(v1, wscale, wqT);
    stats_partial_kernel<<<statsGrid, 128, 0, stream>>>(X, psum, psumsq);
    stats_final_kernel<<<BB * CC / 256, 256, 0, stream>>>(psum, psumsq, meanB, rstdB);
    adain_snake_kernel<<<elemBlocks, 256, 0, stream>>>(
        X, meanB, rstdB, styles + (size_t)(i * 2 + 0) * BB * 2 * CC, alpha1 + i * CC, actB);
    conv_wmma_kernel<<<convGrid, 256, 0, stream>>>(actB, wqT, conv1_b + i * CC,
                                                   nullptr, Y, zeroP, dils[i]);

    // --- adain(Y) -> snake -> conv2 (dil=1) -> + residual X -> X
    wnorm_kernel<<<CC / 256, 256, 0, stream>>>(v2, conv2_g + i * CC, wscale);
    wconvert_kernel<<<KK * CC * CC / 256, 256, 0, stream>>>(v2, wscale, wqT);
    stats_partial_kernel<<<statsGrid, 128, 0, stream>>>(Y, psum, psumsq);
    stats_final_kernel<<<BB * CC / 256, 256, 0, stream>>>(psum, psumsq, meanB, rstdB);
    adain_snake_kernel<<<elemBlocks, 256, 0, stream>>>(
        Y, meanB, rstdB, styles + (size_t)(i * 2 + 1) * BB * 2 * CC, alpha2 + i * CC, actB);
    conv_wmma_kernel<<<convGrid, 256, 0, stream>>>(actB, wqT, conv2_b + i * CC,
                                                   X, X, zeroP, 1);
  }
}
